// FusionSAGEConv_37563783971094
// MI455X (gfx1250) — matrix operand order repaired and checked by reading, not compiled, not run on hardware
//
#include <hip/hip_runtime.h>

// ---------------------------------------------------------------------------
// FusionSAGEConv for MI455X (gfx1250, wave32)
//   out = (mean_{e: dst=v} feat[src_e]) @ W_neigh^T + feat @ W_self^T + b_self
//
// Phase 0: zero accumulators (vectorized b128 stores).
// Phase 1: convert W_self/W_neigh to bf16 once into workspace (64 KB, L2-hot).
// Phase 2: edge scatter, one wave32 per edge, b128 gather + f32 atomics
//          (51 MB target lives in the 192 MB L2 -> L2 atomic units).
// Phase 3: fused mean + dual GEMM on v_wmma_f32_16x16x32_bf16.
//          A tiles staged once per block into LDS (bf16, bank-padded) and
//          shared by all 8 waves; B fragments are raw bf16 global b128 loads.
//          Epilogue uses a uniform in-range fast path (no per-lane branching).
// ---------------------------------------------------------------------------

typedef __attribute__((ext_vector_type(16))) __bf16 v16bf;
typedef __attribute__((ext_vector_type(8)))  __bf16 v8bf;
typedef __attribute__((ext_vector_type(4)))  __bf16 v4bf;
typedef __attribute__((ext_vector_type(8)))  float  v8f;
typedef __attribute__((ext_vector_type(4)))  float  v4f;

#define FEATS      128   // IN_FEATS == OUT_FEATS
#define LDS_STRIDE 136   // 128 + 8 bf16 pad (272B rows) -> spread LDS banks

// ---------------------------------------------------------------- phase 0 --
__global__ void sage_zero_kernel(float* __restrict__ p, long long n) {
  long long i = (long long)blockIdx.x * blockDim.x + threadIdx.x;
  long long n4 = n >> 2;
  if (i < n4) ((v4f*)p)[i] = (v4f){0.f, 0.f, 0.f, 0.f};
  if (i == 0) for (long long r = n4 << 2; r < n; ++r) p[r] = 0.f;
}

// ---------------------------------------------------------------- phase 1 --
// One-time f32 -> bf16 conversion of both weight matrices into workspace.
__global__ void sage_wconv_kernel(const float* __restrict__ Wn,
                                  const float* __restrict__ Ws,
                                  __bf16* __restrict__ wbf) {
  int i = blockIdx.x * blockDim.x + threadIdx.x;   // one float4 per thread
  const int n4 = FEATS * FEATS / 4;                // 4096
  if (i >= n4) return;
  v4f s = ((const v4f*)Ws)[i];
  v4f n = ((const v4f*)Wn)[i];
  v4bf sb, nb;
#pragma unroll
  for (int j = 0; j < 4; ++j) { sb[j] = (__bf16)s[j]; nb[j] = (__bf16)n[j]; }
  ((v4bf*)wbf)[i] = sb;                            // slot 0: W_self
  ((v4bf*)(wbf + FEATS * FEATS))[i] = nb;          // slot 1: W_neigh
}

// ---------------------------------------------------------------- phase 2 --
__global__ void sage_scatter_kernel(const float* __restrict__ feat,
                                    const int*   __restrict__ src,
                                    const int*   __restrict__ dst,
                                    float* __restrict__ summed,
                                    float* __restrict__ deg,
                                    int n_edges) {
  long long tid = (long long)blockIdx.x * blockDim.x + threadIdx.x;
  int edge = (int)(tid >> 5);
  int lane = (int)(tid & 31);
  if (edge >= n_edges) return;

  int s = src[edge];
  int d = dst[edge];

  v4f v = *(const v4f*)(feat + (long long)s * FEATS + lane * 4);
  float* o = summed + (long long)d * FEATS + lane * 4;
  atomicAdd(o + 0, v[0]);
  atomicAdd(o + 1, v[1]);
  atomicAdd(o + 2, v[2]);
  atomicAdd(o + 3, v[3]);
  if (lane == 0) atomicAdd(deg + d, 1.0f);
}

// ---------------------------------------------------------------- phase 3 --
// A-matrix 16x32 bf16 fragment (contiguous k order, same as the B loads): lane holds row (lane&15);
// elems 0..7  -> K = kb + khalf*16 .. +7     (contiguous 16B in LDS)
// elems 8..15 -> K = kb + khalf*16 + 8 .. +15 (contiguous 16B in LDS)
__device__ __forceinline__ v16bf lds_a_frag(const __bf16* rowp, int kb, int khalf) {
  v8bf lo = *(const v8bf*)(rowp + kb + khalf * 16);
  v8bf hi = *(const v8bf*)(rowp + kb + khalf * 16 + 8);
  v16bf a;
#pragma unroll
  for (int i = 0; i < 8; ++i) { a[i] = lo[i]; a[i + 8] = hi[i]; }
  return a;
}

__global__ void __launch_bounds__(256)
sage_gemm_kernel(const float* __restrict__ feat,
                 const float* __restrict__ summed,
                 const float* __restrict__ deg,
                 const __bf16* __restrict__ wbf_self,
                 const __bf16* __restrict__ wbf_neigh,
                 const float* __restrict__ b_self,
                 float* __restrict__ out,
                 int n_nodes) {
  __shared__ __align__(16) __bf16 ldsA[2][16][LDS_STRIDE]; // [self|mean][row][K]

  const int t  = threadIdx.x;
  const int m0 = blockIdx.x * 16;

  // --- cooperative stage: 256 threads; thread t -> row t/16, 8-float K-chunk.
  {
    int row = t >> 4;
    int kc  = (t & 15) * 8;
    int gr  = m0 + row;
    if (gr >= n_nodes) gr = n_nodes - 1;       // clamp (no predication)
    float scale = __builtin_amdgcn_rcpf(fmaxf(deg[gr], 1.0f)); // mean, ~1ulp
    const v4f* f0 = (const v4f*)(feat   + (long long)gr * FEATS + kc);
    const v4f* s0 = (const v4f*)(summed + (long long)gr * FEATS + kc);
    v4f fa = f0[0], fb = f0[1], sa = s0[0], sb = s0[1];
    v8bf pf, ps;
#pragma unroll
    for (int i = 0; i < 4; ++i) {
      pf[i]     = (__bf16)fa[i];
      pf[i + 4] = (__bf16)fb[i];
      ps[i]     = (__bf16)(sa[i] * scale);
      ps[i + 4] = (__bf16)(sb[i] * scale);
    }
    *(v8bf*)&ldsA[0][row][kc] = pf;
    *(v8bf*)&ldsA[1][row][kc] = ps;
  }
  __syncthreads();

  // --- per-wave 16x16 output tile: wave w -> columns [16w, 16w+16).
  const int lane  = t & 31;
  const int wave  = t >> 5;
  const int khalf = lane >> 4;
  const int ncol  = wave * 16 + (lane & 15);

  v8f c;
  {
    float bias = b_self[ncol];
#pragma unroll
    for (int v = 0; v < 8; ++v) c[v] = bias;
  }

  const __bf16* arow_f = &ldsA[0][lane & 15][0];
  const __bf16* arow_n = &ldsA[1][lane & 15][0];
  const __bf16* ws_row = wbf_self  + (long long)ncol * FEATS;
  const __bf16* wn_row = wbf_neigh + (long long)ncol * FEATS;

#pragma unroll
  for (int ks = 0; ks < 4; ++ks) {             // K = 128 in 4 steps of 32
    int kb = ks * 32;
    v16bf a_s = lds_a_frag(arow_f, kb, khalf);
    v16bf b_s = *(const v16bf*)(ws_row + kb + khalf * 16);  // 32B, 2x b128
    c = __builtin_amdgcn_wmma_f32_16x16x32_bf16(false, a_s, false, b_s,
                                                (short)0, c, false, false);
    v16bf a_n = lds_a_frag(arow_n, kb, khalf);
    v16bf b_n = *(const v16bf*)(wn_row + kb + khalf * 16);
    c = __builtin_amdgcn_wmma_f32_16x16x32_bf16(false, a_n, false, b_n,
                                                (short)0, c, false, false);
  }

  // D layout: VGPR v -> M = v + 8*khalf, N = lane&15 inside the wave's tile.
  // Uniform fast path: whole 16-row tile in range (always true when N%16==0),
  // so the bounds test is one scalar branch, not 8 per-lane exec-mask flips.
  const int rbase = m0 + khalf * 8;
  float* obase = out + (long long)rbase * FEATS + ncol;
  if (m0 + 16 <= n_nodes) {
#pragma unroll
    for (int v = 0; v < 8; ++v) obase[v * FEATS] = c[v];
  } else {
#pragma unroll
    for (int v = 0; v < 8; ++v)
      if (rbase + v < n_nodes) obase[v * FEATS] = c[v];
  }
}

// ---------------------------------------------------------------------------
extern "C" void kernel_launch(void* const* d_in, const int* in_sizes, int n_in,
                              void* d_out, int out_size, void* d_ws, size_t ws_size,
                              hipStream_t stream) {
  const float* feat    = (const float*)d_in[0];
  const int*   src     = (const int*)  d_in[1];
  const int*   dst     = (const int*)  d_in[2];
  const float* W_neigh = (const float*)d_in[3];
  const float* W_self  = (const float*)d_in[4];
  const float* b_self  = (const float*)d_in[5];

  const int n_nodes = in_sizes[0] / FEATS;   // 100000
  const int n_edges = in_sizes[1];           // 800000

  // workspace: summed[N,128] f32 | deg[N] f32 | Wbf[2,128,128] bf16
  float*  summed    = (float*)d_ws;
  float*  deg       = summed + (long long)n_nodes * FEATS;
  __bf16* wbf       = (__bf16*)(deg + n_nodes);
  __bf16* wbf_self  = wbf;
  __bf16* wbf_neigh = wbf + FEATS * FEATS;

  long long zn = (long long)n_nodes * FEATS + n_nodes;
  sage_zero_kernel<<<(int)(((zn >> 2) + 255) / 256), 256, 0, stream>>>(summed, zn);

  sage_wconv_kernel<<<(FEATS * FEATS / 4 + 255) / 256, 256, 0, stream>>>(
      W_neigh, W_self, wbf);

  long long sthreads = (long long)n_edges * 32;      // one wave per edge
  sage_scatter_kernel<<<(int)((sthreads + 255) / 256), 256, 0, stream>>>(
      feat, src, dst, summed, deg, n_edges);

  int ntiles = (n_nodes + 15) / 16;                  // 6250 blocks
  sage_gemm_kernel<<<ntiles, 256, 0, stream>>>(
      feat, summed, deg, wbf_self, wbf_neigh, b_self, (float*)d_out, n_nodes);
}
